// DigitalTwinSSM_52037823758466
// MI455X (gfx1250) — compile-verified, matrix operand order checked by reading, not run on hardware
//
#include <hip/hip_runtime.h>
#include <hip/hip_bf16.h>

// Problem constants (from reference)
#define BQ   8
#define TQ   1024
#define NTOK (BQ * TQ)     // 8192 tokens
#define HID  512
#define DIN  1024          // d_inner
#define DST  16            // d_state
#define DTR  32            // dt_rank
#define KIN  128           // padded TOTAL_IN (104 -> 128)
#define NHEAD 128          // padded head cols (32+32+8 -> 128, multiple of 64)

typedef __attribute__((ext_vector_type(16))) __bf16 v16bf;
typedef __attribute__((ext_vector_type(8)))  __bf16 v8bf;
typedef __attribute__((ext_vector_type(8)))  float  v8f;

__device__ __forceinline__ float siluf(float x)     { return x / (1.f + __expf(-x)); }
__device__ __forceinline__ float softplusf(float x) { return (x > 20.f) ? x : log1pf(__expf(x)); }

// ---------------------------------------------------------------------------
// bf16 WMMA GEMM: C[M,N] = act(A[M,K] @ B[K,N] + bias)
// Requirements (guaranteed by caller): M%64==0, N%64==0, K%32==0, pointers
// and leading dims 16B-aligned (lda/ldb/ldc multiples of 8).
// Block = 128 threads (4 wave32). Block tile 64(M) x 64(N), K-step 32.
// Each wave owns 16 rows and computes four 16x16 WMMA tiles per K-step.
// LDS is double-buffered: one __syncthreads per K-step; the next tile's
// global_load_b128s issue before the current step's WMMAs so memory latency
// hides behind the matrix pipe. A is staged in fragment order with 16B
// copies; B is loaded contiguously and transpose-scattered into fragment
// order so every fragment read is 2x ds_load_b128.
// ---------------------------------------------------------------------------
__global__ void k_gemm_bf16(const __bf16* __restrict__ A, const __bf16* __restrict__ Bm,
                            const float* __restrict__ bias,
                            float* __restrict__ C, __bf16* __restrict__ Cbf,
                            int M, int N, int K, int lda, int ldb, int ldc, int act)
{
    __shared__ __align__(32) __bf16 Af[2][4][32][16];   // [buf][wave][lane][frag elem]
    __shared__ __align__(32) __bf16 Bf[2][4][32][16];   // [buf][n-subtile][lane][frag elem]

    const int tid  = threadIdx.x;
    const int wv   = tid >> 5;
    const int lane = tid & 31;
    const int half = lane >> 4;
    const int blockRow = blockIdx.x * 64;
    const int blockCol = blockIdx.y * 64;

    // A: chunk c in [0,256): w=c>>6, ln=(c&63)>>1, hfrag=c&1 -> one 16B copy.
    auto stageA = [&](int buf, int k0, int c) {
        int w = c >> 6, rem = c & 63, ln = rem >> 1, hf = rem & 1;
        const __bf16* src = A + (size_t)(blockRow + w * 16 + (ln & 15)) * lda
                              + (k0 + hf * 16 + (ln >> 4) * 8);
        *reinterpret_cast<v8bf*>(&Af[buf][w][ln][hf * 8]) =
            *reinterpret_cast<const v8bf*>(src);
        if (k0 + 32 < K) __builtin_prefetch(src + 32, 0, 3);
    };
    // B: chunk c in [0,256): nb=c>>6, r=(c&63)>>1, colhalf=c&1 -> contiguous
    // 16B global load, transpose-scatter into fragment order.
    auto stageB = [&](int buf, int k0, int c) {
        int nb = c >> 6, rem = c & 63, r = rem >> 1, ch = rem & 1;
        int hsel, idx;
        if (r < 16) { hsel = r >> 3;        idx = r & 7; }
        else        { hsel = (r - 16) >> 3; idx = 8 + (r & 7); }
        const __bf16* src = Bm + (size_t)(k0 + r) * ldb + (blockCol + nb * 16 + ch * 8);
        v8bf vals = *reinterpret_cast<const v8bf*>(src);
        #pragma unroll
        for (int e = 0; e < 8; ++e)
            Bf[buf][nb][hsel * 16 + ch * 8 + e][idx] = vals[e];
    };
    auto stage = [&](int buf, int k0) {
        stageA(buf, k0, tid);
        stageA(buf, k0, tid + 128);
        stageB(buf, k0, tid);
        stageB(buf, k0, tid + 128);
    };

    v8f acc0 = {}, acc1 = {}, acc2 = {}, acc3 = {};

    stage(0, 0);
    int buf = 0;
    for (int k0 = 0; k0 < K; k0 += 32) {
        __syncthreads();                       // staging of `buf` complete
        if (k0 + 32 < K) stage(buf ^ 1, k0 + 32);   // overlap next tile's loads

        v16bf av = *reinterpret_cast<const v16bf*>(&Af[buf][wv][lane][0]);
        v16bf b0 = *reinterpret_cast<const v16bf*>(&Bf[buf][0][lane][0]);
        v16bf b1 = *reinterpret_cast<const v16bf*>(&Bf[buf][1][lane][0]);
        v16bf b2 = *reinterpret_cast<const v16bf*>(&Bf[buf][2][lane][0]);
        v16bf b3 = *reinterpret_cast<const v16bf*>(&Bf[buf][3][lane][0]);
        acc0 = __builtin_amdgcn_wmma_f32_16x16x32_bf16(false, av, false, b0,
                                                       (short)0, acc0, false, false);
        acc1 = __builtin_amdgcn_wmma_f32_16x16x32_bf16(false, av, false, b1,
                                                       (short)0, acc1, false, false);
        acc2 = __builtin_amdgcn_wmma_f32_16x16x32_bf16(false, av, false, b2,
                                                       (short)0, acc2, false, false);
        acc3 = __builtin_amdgcn_wmma_f32_16x16x32_bf16(false, av, false, b3,
                                                       (short)0, acc3, false, false);
        buf ^= 1;
    }

    // Epilogue: C/D layout — VGPR j: lanes 0-15 -> M=j, lanes 16-31 -> M=j+8
    const int colb = blockCol + (lane & 15);
    float bi0 = bias ? bias[colb]      : 0.f;
    float bi1 = bias ? bias[colb + 16] : 0.f;
    float bi2 = bias ? bias[colb + 32] : 0.f;
    float bi3 = bias ? bias[colb + 48] : 0.f;
    #pragma unroll
    for (int j = 0; j < 8; ++j) {
        const int row = blockRow + wv * 16 + j + 8 * half;
        float v0 = acc0[j] + bi0;
        float v1 = acc1[j] + bi1;
        float v2 = acc2[j] + bi2;
        float v3 = acc3[j] + bi3;
        if (act == 1) {
            v0 = softplusf(v0); v1 = softplusf(v1);
            v2 = softplusf(v2); v3 = softplusf(v3);
        }
        size_t rb = (size_t)row * ldc + colb;
        if (C) {
            C[rb] = v0; C[rb + 16] = v1; C[rb + 32] = v2; C[rb + 48] = v3;
        }
        if (Cbf) {
            Cbf[rb] = (__bf16)v0; Cbf[rb + 16] = (__bf16)v1;
            Cbf[rb + 32] = (__bf16)v2; Cbf[rb + 48] = (__bf16)v3;
        }
    }
}

// ---------------------------------------------------------------------------
// Input pack: concat(x, mask, dt, a) -> bf16, zero-pad 104 -> 128 cols
// ---------------------------------------------------------------------------
__global__ void k_pack_input(const float* __restrict__ x, const float* __restrict__ msk,
                             const float* __restrict__ dts, const float* __restrict__ a,
                             __bf16* __restrict__ dst)
{
    int idx = blockIdx.x * 256 + threadIdx.x;
    if (idx >= NTOK * KIN) return;
    int n = idx >> 7, c = idx & 127;
    float v = 0.f;
    if      (c < 32)  v = x  [n * 32 + c];
    else if (c < 64)  v = msk[n * 32 + (c - 32)];
    else if (c < 96)  v = dts[n * 32 + (c - 64)];
    else if (c < 104) v = a  [n * 8  + (c - 96)];
    dst[idx] = (__bf16)v;
}

// f32 -> bf16 weight convert with optional row zero-padding
__global__ void k_cvt_pad(const float* __restrict__ src, __bf16* __restrict__ dst,
                          int rows, int dstRows, int cols)
{
    int idx = blockIdx.x * 256 + threadIdx.x;
    if (idx >= dstRows * cols) return;
    int r = idx / cols, c = idx % cols;
    dst[idx] = (r < rows) ? (__bf16)src[r * cols + c] : (__bf16)0.f;
}

// Pack mean/logvar/hawkes weights into one 512 x 128 bf16 matrix (cols 72..127 zero)
__global__ void k_pack_heads(const float* __restrict__ mw, const float* __restrict__ lw,
                             const float* __restrict__ hw, __bf16* __restrict__ dst)
{
    int idx = blockIdx.x * 256 + threadIdx.x;
    if (idx >= HID * NHEAD) return;
    int r = idx / NHEAD, c = idx % NHEAD;
    float v = 0.f;
    if      (c < 32) v = mw[r * 32 + c];
    else if (c < 64) v = lw[r * 32 + (c - 32)];
    else if (c < 72) v = hw[r * 8  + (c - 64)];
    dst[idx] = (__bf16)v;
}

// ---------------------------------------------------------------------------
// Depthwise causal conv1d (D_CONV=4) over time + SiLU; writes f32 + bf16
// ---------------------------------------------------------------------------
__global__ void k_conv_silu(const float* __restrict__ xz, const float* __restrict__ cw,
                            const float* __restrict__ cb,
                            float* __restrict__ uf, __bf16* __restrict__ ub)
{
    int idx = blockIdx.x * 256 + threadIdx.x;
    if (idx >= NTOK * DIN) return;
    int n = idx >> 10, d = idx & 1023;
    int b = n >> 10,  t = n & 1023;
    float s = cb[d];
    #pragma unroll
    for (int j = 0; j < 4; ++j) {
        int tp = t - 3 + j;
        if (tp >= 0) s += xz[((size_t)(b * TQ + tp)) * (2 * DIN) + d] * cw[d * 4 + j];
    }
    float u = siluf(s);
    uf[idx] = u;
    ub[idx] = (__bf16)u;
}

// ---------------------------------------------------------------------------
// Selective scan: one thread per (b,d) channel; 16 states in registers.
// Fuses +u*D, gating by silu(z), and bf16 conversion for out_proj.
// B/C state vectors fetched as float4 (global_load_b128).
// ---------------------------------------------------------------------------
__global__ void k_scan(const float* __restrict__ dtf, const float* __restrict__ uf,
                       const float* __restrict__ xz,  const float* __restrict__ xdbl,
                       const float* __restrict__ A_log, const float* __restrict__ Dp,
                       __bf16* __restrict__ ybf)
{
    int d = blockIdx.x * 256 + threadIdx.x;
    int b = blockIdx.y;
    float Ar[DST], st[DST];
    #pragma unroll
    for (int n = 0; n < DST; ++n) { Ar[n] = -__expf(A_log[d * DST + n]); st[n] = 0.f; }
    const float Dd = Dp[d];

    for (int t = 0; t < TQ; ++t) {
        size_t base = (size_t)(b * TQ + t);
        float dtv = dtf[base * DIN + d];
        float uv  = uf [base * DIN + d];
        float zv  = xz [base * (2 * DIN) + DIN + d];

        const float4* xb = reinterpret_cast<const float4*>(xdbl + base * 64 + DTR);
        float Bv[DST], Cv[DST];
        #pragma unroll
        for (int q = 0; q < 4; ++q) {
            float4 tb = xb[q];
            Bv[4 * q + 0] = tb.x; Bv[4 * q + 1] = tb.y;
            Bv[4 * q + 2] = tb.z; Bv[4 * q + 3] = tb.w;
            float4 tc = xb[4 + q];
            Cv[4 * q + 0] = tc.x; Cv[4 * q + 1] = tc.y;
            Cv[4 * q + 2] = tc.z; Cv[4 * q + 3] = tc.w;
        }

        float accv = 0.f;
        #pragma unroll
        for (int n = 0; n < DST; ++n) {
            st[n] = st[n] * __expf(dtv * Ar[n]) + dtv * Bv[n] * uv;
            accv += st[n] * Cv[n];
        }
        float y = (accv + uv * Dd) * siluf(zv);
        ybf[base * DIN + d] = (__bf16)y;
    }
}

// ---------------------------------------------------------------------------
// LayerNorm over HIDDEN=512, one block (256 thr) per token; writes h (f32 to
// d_out) and h bf16 (for heads GEMM / cdsp path).
// ---------------------------------------------------------------------------
__global__ void k_layernorm(const float* __restrict__ X, const float* __restrict__ g,
                            const float* __restrict__ bb,
                            float* __restrict__ Hout, __bf16* __restrict__ Hbf)
{
    __shared__ float red[256];
    int n = blockIdx.x, tid = threadIdx.x;
    const float* xr = X + (size_t)n * HID;
    float v0 = xr[tid], v1 = xr[tid + 256];
    red[tid] = v0 + v1; __syncthreads();
    for (int s = 128; s > 0; s >>= 1) { if (tid < s) red[tid] += red[tid + s]; __syncthreads(); }
    float mu = red[0] * (1.f / HID); __syncthreads();
    float d0 = v0 - mu, d1 = v1 - mu;
    red[tid] = d0 * d0 + d1 * d1; __syncthreads();
    for (int s = 128; s > 0; s >>= 1) { if (tid < s) red[tid] += red[tid + s]; __syncthreads(); }
    float rs = rsqrtf(red[0] * (1.f / HID) + 1e-5f);
    float h0 = d0 * rs * g[tid]       + bb[tid];
    float h1 = d1 * rs * g[tid + 256] + bb[tid + 256];
    Hout[(size_t)n * HID + tid]       = h0;
    Hout[(size_t)n * HID + tid + 256] = h1;
    Hbf [(size_t)n * HID + tid]       = (__bf16)h0;
    Hbf [(size_t)n * HID + tid + 256] = (__bf16)h1;
}

// Heads epilogue: unpack 8192x128 GEMM result into y_pred / y_var / hawkes
__global__ void k_heads(const float* __restrict__ head, const float* __restrict__ mb,
                        const float* __restrict__ lb, const float* __restrict__ hbase,
                        const float* __restrict__ hb,
                        float* __restrict__ ypred, float* __restrict__ yvar,
                        float* __restrict__ hawk)
{
    int idx = blockIdx.x * 256 + threadIdx.x;
    if (idx >= NTOK * 72) return;
    int n = idx / 72, c = idx % 72;
    float v = head[(size_t)n * NHEAD + c];
    if (c < 32)      { ypred[n * 32 + c] = v + mb[c]; }
    else if (c < 64) { int cc = c - 32; yvar[n * 32 + cc] = softplusf(v + lb[cc]) + 1e-4f; }
    else             { int cc = c - 64; hawk[n * 8  + cc] = softplusf(hbase[cc] + v + hb[cc]); }
}

// ---------------------------------------------------------------------------
// cdsp: column stats on h[:, ::4, :] (2048x512) and a[:, ::4, :] (2048x8)
// ---------------------------------------------------------------------------
__global__ void k_stats(const float* __restrict__ H, const float* __restrict__ a_seq,
                        float* __restrict__ hmean, float* __restrict__ hinv,
                        float* __restrict__ amean, float* __restrict__ ainv)
{
    __shared__ float red[256];
    int j = blockIdx.x, tid = threadIdx.x;
    bool ish = (j < HID);
    int c = ish ? j : (j - HID);
    float p = 0.f;
    for (int r = tid; r < 2048; r += 256) {
        int b = r >> 8, tt = (r & 255) * 4;
        size_t tok = (size_t)b * TQ + tt;
        p += ish ? H[tok * HID + c] : a_seq[tok * 8 + c];
    }
    red[tid] = p; __syncthreads();
    for (int s = 128; s > 0; s >>= 1) { if (tid < s) red[tid] += red[tid + s]; __syncthreads(); }
    float mean = red[0] * (1.f / 2048.f); __syncthreads();
    p = 0.f;
    for (int r = tid; r < 2048; r += 256) {
        int b = r >> 8, tt = (r & 255) * 4;
        size_t tok = (size_t)b * TQ + tt;
        float v = ish ? H[tok * HID + c] : a_seq[tok * 8 + c];
        float dd = v - mean;
        p += dd * dd;
    }
    red[tid] = p; __syncthreads();
    for (int s = 128; s > 0; s >>= 1) { if (tid < s) red[tid] += red[tid + s]; __syncthreads(); }
    if (tid == 0) {
        float inv = 1.f / (sqrtf(red[0]) + 1e-6f);
        if (ish) { hmean[c] = mean; hinv[c] = inv; }
        else     { amean[c] = mean; ainv[c] = inv; }
    }
}

__global__ void k_corr(const float* __restrict__ H, const float* __restrict__ a_seq,
                       const float* __restrict__ hmean, const float* __restrict__ hinv,
                       const float* __restrict__ amean, const float* __restrict__ ainv,
                       float* __restrict__ corr)
{
    __shared__ float red[256];
    int c = blockIdx.x, jj = blockIdx.y, tid = threadIdx.x;
    float hm = hmean[c], hi = hinv[c], am = amean[jj], ai = ainv[jj];
    float p = 0.f;
    for (int r = tid; r < 2048; r += 256) {
        int b = r >> 8, tt = (r & 255) * 4;
        size_t tok = (size_t)b * TQ + tt;
        p += (H[tok * HID + c] - hm) * hi * (a_seq[tok * 8 + jj] - am) * ai;
    }
    red[tid] = p; __syncthreads();
    for (int s = 128; s > 0; s >>= 1) { if (tid < s) red[tid] += red[tid + s]; __syncthreads(); }
    if (tid == 0) corr[c * 8 + jj] = red[0];
}

__global__ void k_cdsp(const float* __restrict__ corr, float* __restrict__ out)
{
    __shared__ float red[256];
    int tid = threadIdx.x;
    float p = 0.f;
    for (int i = tid; i < HID * 8; i += 256) { float v = corr[i]; p += v * v; }
    red[tid] = p; __syncthreads();
    for (int s = 128; s > 0; s >>= 1) { if (tid < s) red[tid] += red[tid + s]; __syncthreads(); }
    if (tid == 0) out[0] = red[0] * (1.f / (float)(HID * 8));
}

// ---------------------------------------------------------------------------
extern "C" void kernel_launch(void* const* d_in, const int* in_sizes, int n_in,
                              void* d_out, int out_size, void* d_ws, size_t ws_size,
                              hipStream_t stream)
{
    const float* x_seq     = (const float*)d_in[0];
    const float* mask_seq  = (const float*)d_in[1];
    const float* dt_seq    = (const float*)d_in[2];
    const float* a_seq     = (const float*)d_in[3];
    const float* W_in      = (const float*)d_in[4];
    const float* b_in      = (const float*)d_in[5];
    const float* in_proj_w = (const float*)d_in[6];
    const float* conv_w    = (const float*)d_in[7];
    const float* conv_b    = (const float*)d_in[8];
    const float* x_proj_w  = (const float*)d_in[9];
    const float* dt_proj_w = (const float*)d_in[10];
    const float* dt_proj_b = (const float*)d_in[11];
    const float* A_log     = (const float*)d_in[12];
    const float* D_param   = (const float*)d_in[13];
    const float* out_proj_w= (const float*)d_in[14];
    const float* ln_g      = (const float*)d_in[15];
    const float* ln_b      = (const float*)d_in[16];
    const float* mean_w    = (const float*)d_in[17];
    const float* mean_b    = (const float*)d_in[18];
    const float* logvar_w  = (const float*)d_in[19];
    const float* logvar_b  = (const float*)d_in[20];
    const float* hawkes_base = (const float*)d_in[21];
    const float* hawkes_w  = (const float*)d_in[22];
    const float* hawkes_b  = (const float*)d_in[23];

    float* out = (float*)d_out;
    float* o_ypred = out;                       // 8192*32
    float* o_yvar  = out + NTOK * 32;           // 8192*32
    float* o_hawk  = out + 2 * NTOK * 32;       // 8192*8
    float* o_cdsp  = out + 2 * NTOK * 32 + NTOK * 8;
    float* o_h     = o_cdsp + 1;                // 8192*512

    char* wsb = (char*)d_ws;
    size_t off = 0;
    auto alloc = [&](size_t bytes) -> void* {
        void* p = wsb + off; off += (bytes + 255) & ~(size_t)255; return p;
    };
    __bf16* inp_bf    = (__bf16*)alloc((size_t)NTOK * KIN * 2);
    __bf16* Win_bf    = (__bf16*)alloc((size_t)KIN * HID * 2);
    __bf16* inproj_bf = (__bf16*)alloc((size_t)HID * 2 * DIN * 2);
    __bf16* xproj_bf  = (__bf16*)alloc((size_t)DIN * 64 * 2);
    __bf16* dtproj_bf = (__bf16*)alloc((size_t)DTR * DIN * 2);
    __bf16* outproj_bf= (__bf16*)alloc((size_t)DIN * HID * 2);
    __bf16* headsw_bf = (__bf16*)alloc((size_t)HID * NHEAD * 2);
    __bf16* proj_bf   = (__bf16*)alloc((size_t)NTOK * HID * 2);
    float*  xz_f      = (float*) alloc((size_t)NTOK * 2 * DIN * 4);
    float*  u_f       = (float*) alloc((size_t)NTOK * DIN * 4);
    __bf16* u_bf      = (__bf16*)alloc((size_t)NTOK * DIN * 2);
    float*  xdbl_f    = (float*) alloc((size_t)NTOK * 64 * 4);
    __bf16* xdbl_bf   = (__bf16*)alloc((size_t)NTOK * 64 * 2);
    float*  dt_f      = (float*) alloc((size_t)NTOK * DIN * 4);
    __bf16* y_bf      = (__bf16*)alloc((size_t)NTOK * DIN * 2);
    float*  mout_f    = (float*) alloc((size_t)NTOK * HID * 4);
    __bf16* h_bf      = (__bf16*)alloc((size_t)NTOK * HID * 2);
    float*  head_f    = (float*) alloc((size_t)NTOK * NHEAD * 4);
    float*  hmean     = (float*) alloc(HID * 4);
    float*  hinv      = (float*) alloc(HID * 4);
    float*  amean     = (float*) alloc(8 * 4);
    float*  ainv      = (float*) alloc(8 * 4);
    float*  corr      = (float*) alloc(HID * 8 * 4);

    // --- stage 0: packing / weight conversion ---
    k_pack_input<<<(NTOK * KIN + 255) / 256, 256, 0, stream>>>(x_seq, mask_seq, dt_seq, a_seq, inp_bf);
    k_cvt_pad<<<(KIN * HID + 255) / 256, 256, 0, stream>>>(W_in, Win_bf, 104, KIN, HID);
    k_cvt_pad<<<(HID * 2 * DIN + 255) / 256, 256, 0, stream>>>(in_proj_w, inproj_bf, HID, HID, 2 * DIN);
    k_cvt_pad<<<(DIN * 64 + 255) / 256, 256, 0, stream>>>(x_proj_w, xproj_bf, DIN, DIN, 64);
    k_cvt_pad<<<(DTR * DIN + 255) / 256, 256, 0, stream>>>(dt_proj_w, dtproj_bf, DTR, DTR, DIN);
    k_cvt_pad<<<(DIN * HID + 255) / 256, 256, 0, stream>>>(out_proj_w, outproj_bf, DIN, DIN, HID);
    k_pack_heads<<<(HID * NHEAD + 255) / 256, 256, 0, stream>>>(mean_w, logvar_w, hawkes_w, headsw_bf);

    // --- GEMM 1: proj = inp @ W_in + b_in  (bf16 out only) ---
    k_gemm_bf16<<<dim3(NTOK / 64, HID / 64), 128, 0, stream>>>(
        inp_bf, Win_bf, b_in, nullptr, proj_bf, NTOK, HID, KIN, KIN, HID, HID, 0);

    // --- GEMM 2: xz = proj @ in_proj_w ---
    k_gemm_bf16<<<dim3(NTOK / 64, (2 * DIN) / 64), 128, 0, stream>>>(
        proj_bf, inproj_bf, nullptr, xz_f, nullptr, NTOK, 2 * DIN, HID, HID, 2 * DIN, 2 * DIN, 0);

    // --- conv1d + SiLU ---
    k_conv_silu<<<(NTOK * DIN + 255) / 256, 256, 0, stream>>>(xz_f, conv_w, conv_b, u_f, u_bf);

    // --- GEMM 3: x_dbl = u @ x_proj_w ---
    k_gemm_bf16<<<dim3(NTOK / 64, 64 / 64), 128, 0, stream>>>(
        u_bf, xproj_bf, nullptr, xdbl_f, xdbl_bf, NTOK, 64, DIN, DIN, 64, 64, 0);

    // --- GEMM 4: dt = softplus(dt_r @ dt_proj_w + dt_proj_b)  (A = first 32 cols of x_dbl) ---
    k_gemm_bf16<<<dim3(NTOK / 64, DIN / 64), 128, 0, stream>>>(
        xdbl_bf, dtproj_bf, dt_proj_b, dt_f, nullptr, NTOK, DIN, DTR, 64, DIN, DIN, 1);

    // --- selective scan (fused +u*D, *silu(z), bf16 out) ---
    k_scan<<<dim3(DIN / 256, BQ), 256, 0, stream>>>(dt_f, u_f, xz_f, xdbl_f, A_log, D_param, y_bf);

    // --- GEMM 5: mamba_out = y @ out_proj_w ---
    k_gemm_bf16<<<dim3(NTOK / 64, HID / 64), 128, 0, stream>>>(
        y_bf, outproj_bf, nullptr, mout_f, nullptr, NTOK, HID, DIN, DIN, HID, HID, 0);

    // --- layernorm -> h (f32 to d_out) + h bf16 ---
    k_layernorm<<<NTOK, 256, 0, stream>>>(mout_f, ln_g, ln_b, o_h, h_bf);

    // --- GEMM 6: heads = h @ [mean|logvar|hawkes] ---
    k_gemm_bf16<<<dim3(NTOK / 64, NHEAD / 64), 128, 0, stream>>>(
        h_bf, headsw_bf, nullptr, head_f, nullptr, NTOK, NHEAD, HID, HID, NHEAD, NHEAD, 0);

    k_heads<<<(NTOK * 72 + 255) / 256, 256, 0, stream>>>(
        head_f, mean_b, logvar_b, hawkes_base, hawkes_b, o_ypred, o_yvar, o_hawk);

    // --- cdsp ---
    k_stats<<<HID + 8, 256, 0, stream>>>(o_h, a_seq, hmean, hinv, amean, ainv);
    k_corr<<<dim3(HID, 8), 256, 0, stream>>>(o_h, a_seq, hmean, hinv, amean, ainv, corr);
    k_cdsp<<<1, 256, 0, stream>>>(corr, o_cdsp);
}
